// STGCNBlock_11038065951432
// MI455X (gfx1250) — matrix operand order repaired and checked by reading, not compiled, not run on hardware
//
#include <hip/hip_runtime.h>

typedef __bf16 bf16;
typedef __attribute__((ext_vector_type(16))) __bf16 v16bf;
typedef __attribute__((ext_vector_type(8)))  float  v8f;
typedef __attribute__((ext_vector_type(2)))  float  f2v;
typedef __attribute__((ext_vector_type(2)))  __bf16 bf2v;

#define NB 64
#define CC 64
#define TDIM 300
#define VV 25
#define PP 3
#define ROWS (NB*TDIM*VV)   /* 480000 */
#define EPSBN 1e-5f

// ---- async global->LDS (CDNA5 GLOBAL_LOAD_ASYNC_TO_LDS_B128, ASYNCcnt) ----
#if defined(__has_builtin)
#if __has_builtin(__builtin_amdgcn_global_load_async_to_lds_b128)
#define ASYNC_LDS 1
#endif
#endif
#ifndef ASYNC_LDS
#define ASYNC_LDS 0
#endif

#if ASYNC_LDS
typedef int v4i __attribute__((vector_size(16)));
typedef __attribute__((address_space(1))) v4i gv4i;   // global (__device__) int4
typedef __attribute__((address_space(3))) v4i lv4i;   // LDS (__shared__) int4
__device__ inline void async_cp16(const void* gsrc, void* ldst){
  __builtin_amdgcn_global_load_async_to_lds_b128((gv4i*)gsrc, (lv4i*)ldst, 0, 0);
}
__device__ inline void async_wait(){
#if __has_builtin(__builtin_amdgcn_s_wait_asynccnt)
  __builtin_amdgcn_s_wait_asynccnt(0);
#else
  asm volatile("s_wait_asynccnt 0x0" ::: "memory");
#endif
}
#endif

union FragU { v16bf v; unsigned u[8]; bf16 h[16]; };

__device__ inline unsigned pack2(float a, float b){
  f2v x; x.x = a; x.y = b;
  union { bf2v v; unsigned u; } t;
  t.v = __builtin_convertvector(x, bf2v);   // -> single v_cvt_pk_bf16_f32
  return t.u;
}

// A-matrix 16x32 bf16 fragment: element(m,k) = p[(r0+m)*ld + k0+k]
__device__ inline v16bf frag_a_ld(const bf16* p, int r0, int k0, int ld){
  int lane = threadIdx.x & 31;
  int row = r0 + (lane & 15);
  int kb  = k0 + ((lane & 16) ? 8 : 0);
  const unsigned* rp = (const unsigned*)(p + row*ld + kb);
  FragU f;
#pragma unroll
  for (int j=0;j<4;++j){ f.u[j] = rp[j]; f.u[4+j] = rp[8+j]; }
  return f.v;
}

// B-matrix KxN(32x16) bf16 fragment: element(k,n) = p[(n0+n)*ld + k0+k]
__device__ inline v16bf frag_b_ld(const bf16* p, int n0, int k0, int ld){
  int lane = threadIdx.x & 31;
  int row = n0 + (lane & 15);
  int kb  = k0 + ((lane & 16) ? 8 : 0);
  const unsigned* rp = (const unsigned*)(p + row*ld + kb);
  FragU f;
#pragma unroll
  for (int j=0;j<4;++j){ f.u[j] = rp[j]; f.u[4+j] = rp[8+j]; }
  return f.v;
}

__device__ inline v8f wmma_bf16(v16bf a, v16bf b, v8f c){
  return __builtin_amdgcn_wmma_f32_16x16x32_bf16(false, a, false, b, (short)0, c, false, false);
}

// ---------------- prep: zero stats, convert weights to bf16 ----------------
__global__ __launch_bounds__(256) void k_prep(const float* __restrict__ A,
                                              const float* __restrict__ E,
                                              const float* __restrict__ wg,
                                              const float* __restrict__ wt,
                                              float* stats, bf16* WgB, bf16* A2B,
                                              float* CS, bf16* WtB){
  int tid = threadIdx.x;
  for (int i=tid;i<256;i+=256) stats[i] = 0.f;                 // s1,q1,s2,q2
  for (int i=tid;i<192*64;i+=256) WgB[i] = (bf16)wg[i];
  for (int i=tid;i<32*96;i+=256) A2B[i] = (bf16)0.f;
  for (int i=tid;i<9*64*64;i+=256){
    int dk = i/4096, rem = i%4096, co = rem>>6, ci = rem&63;
    WtB[i] = (bf16)wt[(co*64+ci)*9 + dk];                      // [dk][co][ci]
  }
  __syncthreads();
  for (int i=tid;i<PP*VV*VV;i+=256){
    int p = i/(VV*VV), r = i%(VV*VV), v = r/VV, w = r%VV;
    A2B[w*96 + p*32 + v] = (bf16)(A[i]*E[i]);                  // A2[w][p*32+v]
  }
  for (int i=tid;i<PP*32;i+=256){
    int p = i>>5, w = i&31; float s = 0.f;
    if (w < VV)
      for (int v=0;v<VV;++v) s += A[(p*VV+v)*VV+w]*E[(p*VV+v)*VV+w];
    CS[i] = s;                                                 // colsum[p*32+w]
  }
}

// ---------------- fused GCN (1x1 conv GEMM + graph aggregation) ----------------
__global__ __launch_bounds__(256) void k_gcn(const float* __restrict__ x,
                                             const float* __restrict__ bgcn,
                                             const bf16* __restrict__ WgG,
                                             const bf16* __restrict__ A2G,
                                             const float* __restrict__ CSG,
                                             float* __restrict__ y,
                                             float* gstats){
  __shared__ bf16 Xl[32][72];     // v x ci (rows 25..31 zero)
  __shared__ bf16 Wgl[192][72];   // o x ci
  __shared__ bf16 A2l[32][104];   // w x kv (kv = p*32+v, K=96)
  __shared__ bf16 Hl[64][104];    // c x kv
  __shared__ float CSs[96];
  __shared__ float bgs[192];
  __shared__ float s1l[64], q1l[64];

  int tid = threadIdx.x, lane = tid & 31, wave = tid >> 5;
  int n  = blockIdx.x / 30;
  int t0 = (blockIdx.x % 30) * 10;

#if ASYNC_LDS
  for (int i=tid;i<192*8;i+=256){ int o=i>>3, j=i&7; async_cp16(WgG + o*64 + j*8, &Wgl[o][0] + j*8); }
  for (int i=tid;i<32*12;i+=256){ int w=i/12, j=i%12; async_cp16(A2G + w*96 + j*8, &A2l[w][0] + j*8); }
#else
  for (int i=tid;i<6144;i+=256){ int o=i>>5, j=i&31; ((unsigned*)&Wgl[o][0])[j] = ((const unsigned*)WgG)[i]; }
  for (int i=tid;i<1536;i+=256){ int w=i/48, j=i%48; ((unsigned*)&A2l[w][0])[j] = ((const unsigned*)A2G)[i]; }
#endif
  for (int i=tid;i<96;i+=256)  CSs[i] = CSG[i];
  for (int i=tid;i<192;i+=256) bgs[i] = bgcn[i];
  if (tid < 64){ s1l[tid]=0.f; q1l[tid]=0.f; }
  for (int i=tid;i<7*64;i+=256){ int v=25+i/64, c=i%64; Xl[v][c] = (bf16)0.f; }
#if ASYNC_LDS
  async_wait();
#endif

  int ct2 = wave & 3, wt2 = wave >> 2;
  int cg = ct2*16 + (lane & 15);   // this lane's output channel in GEMM2
  float lsum = 0.f, lsq = 0.f;

  for (int tt=0; tt<10; ++tt){
    int t = t0 + tt;
    __syncthreads();
    // stage X[v][ci] (bf16) ; v fastest for coalescing
    for (int i=tid;i<2048;i+=256){
      int ci = i>>5, v = i&31;
      if (v < 25) Xl[v][ci] = (bf16)x[((n*64+ci)*300 + t)*25 + v];
    }
    __syncthreads();
    // GEMM1: h[v,o] = X(32x64) * Wg^T ; 2x12 tiles over 8 waves
#pragma unroll
    for (int i=0;i<3;++i){
      int idx = wave*3 + i;
      int mt = idx & 1, nt = idx >> 1;
      v8f acc = {0.f,0.f,0.f,0.f,0.f,0.f,0.f,0.f};
#pragma unroll
      for (int ks=0;ks<2;++ks){
        v16bf a = frag_a_ld(&Xl[0][0],  mt*16, ks*32, 72);
        v16bf b = frag_b_ld(&Wgl[0][0], nt*16, ks*32, 72);
        acc = wmma_bf16(a, b, acc);
      }
      // restripe to Hl[c][p*32+v]
      int nn = lane & 15, o = nt*16 + nn, c = o & 63, p = o >> 6;
      int vb = mt*16 + ((lane & 16) ? 8 : 0);
      unsigned* dst = (unsigned*)&Hl[c][p*32 + vb];
#pragma unroll
      for (int r=0;r<4;++r) dst[r] = pack2(acc[2*r], acc[2*r+1]);
    }
    __syncthreads();
    // GEMM2: y[w,c] = A2(32x96) * H(96x64) ; 2x4 tiles over 8 waves
    v8f acc = {0.f,0.f,0.f,0.f,0.f,0.f,0.f,0.f};
#pragma unroll
    for (int ks=0;ks<3;++ks){
      v16bf a = frag_a_ld(&A2l[0][0], wt2*16, ks*32, 104);
      v16bf b = frag_b_ld(&Hl[0][0],  ct2*16, ks*32, 104);
      acc = wmma_bf16(a, b, acc);
    }
    int mb = (lane & 16) ? 8 : 0;
    float bg0 = bgs[cg], bg1 = bgs[64+cg], bg2 = bgs[128+cg];
#pragma unroll
    for (int r=0;r<8;++r){
      int w = wt2*16 + mb + r;
      if (w < 25){
        float bias = CSs[w]*bg0 + CSs[32+w]*bg1 + CSs[64+w]*bg2;
        float val  = acc[r] + bias;
        y[(size_t)((n*300 + t)*25 + w)*64 + cg] = val;
        lsum += val; lsq += val*val;
      }
    }
  }
  __syncthreads();
  atomicAdd(&s1l[cg], lsum);
  atomicAdd(&q1l[cg], lsq);
  __syncthreads();
  if (tid < 64){ atomicAdd(&gstats[tid], s1l[tid]); atomicAdd(&gstats[64+tid], q1l[tid]); }
}

// ---------------- BN affine solve ----------------
__global__ void k_affine(const float* __restrict__ gamma, const float* __restrict__ beta,
                         float* stats, int stage){
  int c = threadIdx.x;
  if (c < 64){
    float s = stats[stage*128 + c], q = stats[stage*128 + 64 + c];
    float cnt = (float)ROWS;
    float mu = s/cnt, var = q/cnt - mu*mu;
    float a = gamma[c] * rsqrtf(var + EPSBN);
    stats[256 + stage*128 + c]      = a;
    stats[256 + stage*128 + 64 + c] = beta[c] - mu*a;
  }
}

// ---------------- BN1 apply + ReLU -> bf16 activations (128-bit I/O) ----------------
__global__ __launch_bounds__(256) void k_bnrelu(const float4* __restrict__ y4,
                                                const float* __restrict__ stats,
                                                uint4* __restrict__ yrV, int totalVec){
  __shared__ float a1[64], b1[64];
  if (threadIdx.x < 64){ a1[threadIdx.x]=stats[256+threadIdx.x]; b1[threadIdx.x]=stats[320+threadIdx.x]; }
  __syncthreads();
  int i = blockIdx.x*256 + threadIdx.x;
  int stride = gridDim.x*256;
  for (; i < totalVec; i += stride){        // 8 elements per iteration
    float4 p0 = y4[i*2], p1 = y4[i*2+1];
    int c = (i*8) & 63;
    uint4 o;
    o.x = pack2(fmaxf(a1[c  ]*p0.x + b1[c  ], 0.f), fmaxf(a1[c+1]*p0.y + b1[c+1], 0.f));
    o.y = pack2(fmaxf(a1[c+2]*p0.z + b1[c+2], 0.f), fmaxf(a1[c+3]*p0.w + b1[c+3], 0.f));
    o.z = pack2(fmaxf(a1[c+4]*p1.x + b1[c+4], 0.f), fmaxf(a1[c+5]*p1.y + b1[c+5], 0.f));
    o.w = pack2(fmaxf(a1[c+6]*p1.z + b1[c+6], 0.f), fmaxf(a1[c+7]*p1.w + b1[c+7], 0.f));
    yrV[i] = o;
  }
}

// ---------------- TCN: (9,1) conv as 9 row-shifted GEMMs ----------------
__global__ __launch_bounds__(256) void k_tcn(const bf16* __restrict__ yr,
                                             const bf16* __restrict__ WtG,
                                             const float* __restrict__ btcn,
                                             float* __restrict__ z,
                                             float* gstats){
  __shared__ bf16 Yl[328][72];                 // 128-row block + 100-row halo each side
  __shared__ float s2l[64], q2l[64];
  int tid = threadIdx.x, lane = tid & 31, wave = tid >> 5;
  int r0 = blockIdx.x * 128;
  if (tid < 64){ s2l[tid]=0.f; q2l[tid]=0.f; }
#if ASYNC_LDS
  for (int i=tid;i<328*8;i+=256){
    int row = i>>3, j = i&7;                   // 16B chunk j of row
    int g = r0 - 100 + row;
    bf16* dst = &Yl[0][0] + row*72 + j*8;
    if (g >= 0 && g < ROWS){
      async_cp16(yr + (size_t)g*64 + j*8, dst);
    } else {
      unsigned* d4 = (unsigned*)dst;
      d4[0]=0; d4[1]=0; d4[2]=0; d4[3]=0;
    }
  }
  async_wait();
#else
  const unsigned* yrU = (const unsigned*)yr;
  for (int i=tid;i<328*32;i+=256){
    int row = i>>5, j = i&31;
    int g = r0 - 100 + row;
    unsigned v = 0;
    if (g >= 0 && g < ROWS) v = yrU[(size_t)g*32 + j];
    ((unsigned*)&Yl[0][0])[row*36 + j] = v;
  }
#endif
  __syncthreads();

  int mt = wave;                                        // 8 M-tiles over 8 waves
  int rrow = r0 + mt*16 + (lane & 15);
  int tloc = (rrow % 7500) / 25;
  int mb = (lane & 16) ? 8 : 0;

  for (int ct=0; ct<4; ++ct){
    v8f acc = {0.f,0.f,0.f,0.f,0.f,0.f,0.f,0.f};
    for (int dk=0; dk<9; ++dk){
      bool valid = (unsigned)(tloc + dk - 4) < 300u;    // per-lane temporal boundary
      int rb = mt*16 + 100 + (dk-4)*25;                 // local halo base row
#pragma unroll
      for (int ks=0; ks<2; ++ks){
        FragU fa; fa.v = frag_a_ld(&Yl[0][0], rb, ks*32, 72);
        if (!valid){
#pragma unroll
          for (int j=0;j<8;++j) fa.u[j] = 0;
        }
        v16bf b = frag_b_ld(WtG + dk*4096, ct*16, ks*32, 64);  // weights from L2/WGP$
        acc = wmma_bf16(fa.v, b, acc);
      }
    }
    int co = ct*16 + (lane & 15);
    float bt = btcn[co];
    float lsum = 0.f, lsq = 0.f;
#pragma unroll
    for (int r=0;r<8;++r){
      int row = r0 + mt*16 + mb + r;
      float val = acc[r] + bt;
      z[(size_t)row*64 + co] = val;
      lsum += val; lsq += val*val;
    }
    atomicAdd(&s2l[co], lsum); atomicAdd(&q2l[co], lsq);
  }
  __syncthreads();
  if (tid < 64){ atomicAdd(&gstats[128+tid], s2l[tid]); atomicAdd(&gstats[192+tid], q2l[tid]); }
}

// ---------------- BN2 + residual + ReLU, NTVC -> NCTV transpose (128-bit I/O) ----------------
__global__ __launch_bounds__(256) void k_out(const float* __restrict__ z,
                                             const float* __restrict__ x,
                                             const float* __restrict__ stats,
                                             float* __restrict__ out){
  __shared__ float tile[64][65];
  __shared__ float a2s[64], b2s[64];
  int tid = threadIdx.x;
  if (tid < 64){ a2s[tid]=stats[384+tid]; b2s[tid]=stats[448+tid]; }
  int n   = blockIdx.x / 118;
  int tv0 = (blockIdx.x % 118) * 64;
  int m = 7500 - tv0; if (m > 64) m = 64;          // 64 or 12 (both multiples of 4)
  for (int i=tid;i<64*16;i+=256){
    int row = i>>4, c4 = (i&15)*4;
    if (row < m){
      float4 v = *(const float4*)&z[(size_t)(n*7500 + tv0 + row)*64 + c4];
      tile[row][c4]=v.x; tile[row][c4+1]=v.y; tile[row][c4+2]=v.z; tile[row][c4+3]=v.w;
    }
  }
  __syncthreads();
  for (int i=tid;i<64*16;i+=256){
    int c = i>>4, j4 = (i&15)*4;
    if (j4 < m){
      size_t oi = (size_t)(n*64 + c)*7500 + tv0 + j4;
      float4 xv = *(const float4*)&x[oi];
      float a = a2s[c], b = b2s[c];
      float4 ov;
      ov.x = fmaxf(a*tile[j4  ][c] + b + xv.x, 0.f);
      ov.y = fmaxf(a*tile[j4+1][c] + b + xv.y, 0.f);
      ov.z = fmaxf(a*tile[j4+2][c] + b + xv.z, 0.f);
      ov.w = fmaxf(a*tile[j4+3][c] + b + xv.w, 0.f);
      *(float4*)&out[oi] = ov;
    }
  }
}

extern "C" void kernel_launch(void* const* d_in, const int* in_sizes, int n_in,
                              void* d_out, int out_size, void* d_ws, size_t ws_size,
                              hipStream_t stream){
  (void)in_sizes; (void)n_in; (void)out_size; (void)ws_size;
  const float* x   = (const float*)d_in[0];
  const float* A   = (const float*)d_in[1];
  const float* E   = (const float*)d_in[2];
  const float* wg  = (const float*)d_in[3];
  const float* bg  = (const float*)d_in[4];
  const float* g1  = (const float*)d_in[5];
  const float* be1 = (const float*)d_in[6];
  const float* wt  = (const float*)d_in[7];
  const float* bt  = (const float*)d_in[8];
  const float* g2  = (const float*)d_in[9];
  const float* be2 = (const float*)d_in[10];

  char* ws = (char*)d_ws;
  float* stats = (float*)(ws + 0);            // 8*64 f32: s1,q1,s2,q2,a1,b1,a2,b2
  bf16*  WgB   = (bf16*) (ws + 2048);         // 192*64 bf16
  bf16*  A2B   = (bf16*) (ws + 26624);        // 32*96 bf16
  float* CS    = (float*)(ws + 32768);        // 96 f32
  bf16*  WtB   = (bf16*) (ws + 33280);        // 9*64*64 bf16
  bf16*  yr    = (bf16*) (ws + 107520);       // 30.72M bf16
  float* y     = (float*)(ws + 61547520ll);   // 30.72M f32 (z aliases y)
  float* z     = y;
  float* out   = (float*)d_out;

  k_prep  <<<1,    256, 0, stream>>>(A, E, wg, wt, stats, WgB, A2B, CS, WtB);
  k_gcn   <<<1920, 256, 0, stream>>>(x, bg, WgB, A2B, CS, y, stats);
  k_affine<<<1,    64,  0, stream>>>(g1, be1, stats, 0);
  k_bnrelu<<<15000,256, 0, stream>>>((const float4*)y, stats, (uint4*)yr, 3840000);
  k_tcn   <<<3750, 256, 0, stream>>>(yr, WtB, bt, z, stats);
  k_affine<<<1,    64,  0, stream>>>(g2, be2, stats, 1);
  k_out   <<<7552, 256, 0, stream>>>(z, x, stats, out);
}
